// PyCNNTorchModel_47923245088982
// MI455X (gfx1250) — matrix operand order repaired and checked by reading, not compiled, not run on hardware
//
#include <hip/hip_runtime.h>

// ---------------------------------------------------------------------------
// Problem constants
// ---------------------------------------------------------------------------
#define BATCH   64
#define CH      3
#define H       224
#define HW      (H * H)            // 50176
#define POOL    111                // (H-2)/2
#define NFILT   8
#define FEAT    (POOL * POOL * NFILT)   // 98568
#define KPAD    98592              // FEAT rounded up to multiple of 32
#define KITERS  (KPAD / 32)        // 3081
#define NSTREAM 128                // split-K streams for GEMM1
#define N1      512
#define N2      256
#define NOUT    1000
#define NOUTP   1008               // padded to multiple of 16

// ---------------------------------------------------------------------------
// CDNA5 WMMA types (wave32; 16x16x32 bf16 -> f32)
// ---------------------------------------------------------------------------
typedef __attribute__((ext_vector_type(16))) __bf16 v16bf;
typedef __attribute__((ext_vector_type(8)))  __bf16 v8bf;
typedef __attribute__((ext_vector_type(8)))  float  v8f;
typedef __attribute__((ext_vector_type(4))) unsigned int u32x4;
typedef __attribute__((ext_vector_type(4))) int          i32x4;
typedef __attribute__((ext_vector_type(8))) int          i32x8;

#define WMMA_BF16(a, b, c) \
    __builtin_amdgcn_wmma_f32_16x16x32_bf16(false, (a), false, (b), (short)0, (c), false, false)

// A-matrix fragment (16x32 bf16, row-major source with leading dim lda).
// ISA 7.12.2: lane L (half=L>>4, m=L&15) holds k = {half*8+0..7, 16+half*8+0..7}
// => two contiguous 8 x bf16 runs => two 128-bit loads.
static __device__ inline v16bf load_a_bf16(const __bf16* A, int lda,
                                           int mbase, int k0, int lane) {
    const int half = lane >> 4;
    const int m    = lane & 15;
    const __bf16* p = A + (size_t)(mbase + m) * (size_t)lda + (size_t)(k0 + half * 8);
    v8bf lo = *(const v8bf*)(p);
    v8bf hi = *(const v8bf*)(p + 16);
    return __builtin_shufflevector(lo, hi, 0, 1, 2, 3, 4, 5, 6, 7,
                                           8, 9, 10, 11, 12, 13, 14, 15);
}

// Clamped B-fragment loader (used only on the K tail tile and small GEMMs).
static __device__ inline v16bf load_b_f32_as_bf16(const float* B, int ldb,
                                                  int k0, int nbase,
                                                  int lane, int kmax) {
    const int half = lane >> 4;
    const int n    = nbase + (lane & 15);
    v16bf r;
#pragma unroll
    for (int e = 0; e < 16; ++e) {
        int k = k0 + half * 8 + (e & 7) + ((e >> 3) << 4);
        if (k >= kmax) k = kmax - 1;
        r[e] = (__bf16)B[(size_t)k * (size_t)ldb + (size_t)n];
    }
    return r;
}

// ---------------------------------------------------------------------------
// Tensor Data Mover: per-wave 2-D tile load (64 rows x 32 bf16 cols) into LDS.
// D# built per CDNA5 ISA ch.8. Groups 2/3 zero (2-D tensor).
// ---------------------------------------------------------------------------
#if defined(__has_builtin)
#  if __has_builtin(__builtin_amdgcn_tensor_load_to_lds)
#    define HAVE_TDM 1
#  endif
#endif
#if __has_include(<hip/amd_detail/amd_gfx1250_TDM.h>)
#  define TDM_6ARG 1
#endif

static __device__ inline void tdm_load_a_tile(const __bf16* gtile,
                                              unsigned lds_off, int lane) {
#if defined(HAVE_TDM)
    unsigned long long ga = (unsigned long long)(uintptr_t)gtile;
    u32x4 g0;
    g0.x = 1u;                                       // count=1, user D#
    g0.y = lds_off;                                  // lds_addr (bytes)
    g0.z = (unsigned)ga;                             // global_addr[31:0]
    g0.w = ((unsigned)(ga >> 32) & 0x01FFFFFFu) | (2u << 30); // addr[56:32]|type=2
    i32x8 g1;
    g1[0] = 0x00010000;                              // data_size=1 (2 bytes)
    g1[1] = (int)(((unsigned)KPAD & 0xFFFFu) << 16); // tensor_dim0[15:0]
    g1[2] = (int)(((unsigned)KPAD >> 16) | (64u << 16)); // dim0[31:16]|tensor_dim1=64
    g1[3] = (int)(32u << 16);                        // tile_dim0=32
    g1[4] = 64;                                      // tile_dim1=64, tile_dim2=0
    g1[5] = KPAD;                                    // tensor_dim0_stride[31:0]
    g1[6] = 0;
    g1[7] = 0;
    i32x4 z4 = {0, 0, 0, 0};
#  if defined(TDM_6ARG)
    i32x8 z8 = {0, 0, 0, 0, 0, 0, 0, 0};
    __builtin_amdgcn_tensor_load_to_lds(g0, g1, z4, z4, z8, 0);
#  else
    __builtin_amdgcn_tensor_load_to_lds(g0, g1, z4, z4, 0);
#  endif
#else
    // Fallback: cooperative wave copy (4 KB tile = 256 x 16B chunks)
    __bf16* lds = (__bf16*)(uintptr_t)lds_off;
#pragma unroll
    for (int j = 0; j < 8; ++j) {
        int cidx = lane + 32 * j;               // 16B chunk index
        int row  = cidx >> 1, ch = cidx & 1;
        *(v8bf*)((char*)lds + (size_t)cidx * 16) =
            *(const v8bf*)(gtile + (size_t)row * KPAD + ch * 8);
    }
#endif
}

// ---------------------------------------------------------------------------
// Kernel 1: fused channel-sum + 3x3 conv + ReLU + 2x2 maxpool -> bf16 flat A
// ---------------------------------------------------------------------------
__global__ void k_conv_pool(const float* __restrict__ x,
                            const float* __restrict__ flt,
                            __bf16* __restrict__ A) {
    int idx = blockIdx.x * blockDim.x + threadIdx.x;      // < BATCH*KPAD
    int b    = idx / KPAD;
    int feat = idx - b * KPAD;
    if (b >= BATCH) return;
    if (feat >= FEAT) { A[(size_t)b * KPAD + feat] = (__bf16)0.0f; return; }

    int f   = feat / (POOL * POOL);
    int rem = feat - f * (POOL * POOL);
    int i   = rem / POOL;
    int j   = rem - i * POOL;

    float w[9];
#pragma unroll
    for (int t = 0; t < 9; ++t) w[t] = flt[f * 9 + t];

    const float* xb = x + (size_t)b * CH * HW;
    float s[4][4];
#pragma unroll
    for (int r = 0; r < 4; ++r) {
#pragma unroll
        for (int c = 0; c < 4; ++c) {
            size_t o = (size_t)(2 * i + r) * H + (size_t)(2 * j + c);
            s[r][c] = xb[o] + xb[o + HW] + xb[o + 2 * HW];
        }
    }

    float m = -3.402823466e38f;
#pragma unroll
    for (int pr = 0; pr < 2; ++pr) {
#pragma unroll
        for (int pc = 0; pc < 2; ++pc) {
            float acc = 0.f;
#pragma unroll
            for (int dy = 0; dy < 3; ++dy)
#pragma unroll
                for (int dx = 0; dx < 3; ++dx)
                    acc = fmaf(s[pr + dy][pc + dx], w[dy * 3 + dx], acc);
            m = fmaxf(m, acc);
        }
    }
    A[(size_t)b * KPAD + feat] = (__bf16)fmaxf(m, 0.0f);
}

// ---------------------------------------------------------------------------
// Kernel 2: zero the f32 split-K accumulator (64x512)
// ---------------------------------------------------------------------------
__global__ void k_zero(float* __restrict__ p, int n) {
    int i = blockIdx.x * blockDim.x + threadIdx.x;
    if (i < n) p[i] = 0.0f;
}

// ---------------------------------------------------------------------------
// Kernel 3: GEMM1  acc(64x512) += A(64xKPAD bf16) @ W1(f32->bf16 in regs)
// grid = (16 N-tiles of 32 cols, 16 K-groups), block = 8 waves -> 128 streams.
// Each wave: 32-col N-tile (full 128B W1 lines) x all four M-tiles.
// A slabs staged by the Tensor Data Mover (double-buffered, s_wait_tensorcnt);
// W1 streamed once with NT loads (202MB single pass; protect L2) + prefetch.
// Clamped addressing only on the single K-tail tile (uniform branch).
// ---------------------------------------------------------------------------
__global__ void __launch_bounds__(256)
k_gemm1(const __bf16* __restrict__ A, const float* __restrict__ W1,
        float* __restrict__ acc) {
    __shared__ __align__(16) __bf16 ldsA[8][2][64 * 32];   // 64 KB

    const int lane  = threadIdx.x & 31;
    const int half  = lane >> 4;
    const int wv    = __builtin_amdgcn_readfirstlane(threadIdx.x >> 5); // 0..7
    const int nbase = blockIdx.x * 32;
    const int sid   = blockIdx.y * 8 + wv;          // k-stream 0..127
    const int n     = nbase + (lane & 15);

    v8f c[4][2];
#pragma unroll
    for (int t = 0; t < 4; ++t)
#pragma unroll
        for (int u = 0; u < 2; ++u)
            c[t][u] = (v8f){0.f, 0.f, 0.f, 0.f, 0.f, 0.f, 0.f, 0.f};

    // Per-lane invariant base for the fast B path: row (half*8), column n.
    const float* bbase = W1 + (size_t)(half * 8) * N1 + n;

    // Prologue: TDM-load the first A slab for this wave's k-stream.
    tdm_load_a_tile(A + (size_t)sid * 32,
                    (unsigned)(uintptr_t)&ldsA[wv][0][0], lane);

    int buf = 0;
    for (int ki = sid; ki < KITERS; ki += NSTREAM) {
        const int k0 = ki * 32;
        const int kn = ki + NSTREAM;

        if (kn < KITERS) {
            // Double-buffer: kick off next A slab, prefetch next W1 slab.
            tdm_load_a_tile(A + (size_t)kn * 32,
                            (unsigned)(uintptr_t)&ldsA[wv][buf ^ 1][0], lane);
            __builtin_prefetch(bbase + (size_t)(kn * 32) * N1, 0, 1);
        }

        // B fragments for both 16-col subtiles (f32 stream -> bf16 regs).
        v16bf b0, b1;
        if (ki != KITERS - 1) {
            // Fast path: one base pointer, constant immediate offsets,
            // non-temporal (W1 is a 202MB single-pass stream).
            const float* bp = bbase + (size_t)k0 * N1;
#pragma unroll
            for (int e = 0; e < 16; ++e) {
                b0[e] = (__bf16)__builtin_nontemporal_load(bp + (size_t)((e & 7) + ((e >> 3) << 4)) * N1);
                b1[e] = (__bf16)__builtin_nontemporal_load(bp + (size_t)((e & 7) + ((e >> 3) << 4)) * N1 + 16);
            }
        } else {
            // Tail tile (k 98560..98591): clamp reads; zero A-pad nullifies.
            b0 = load_b_f32_as_bf16(W1, N1, k0, nbase,      lane, FEAT);
            b1 = load_b_f32_as_bf16(W1, N1, k0, nbase + 16, lane, FEAT);
        }

        if (kn < KITERS) __builtin_amdgcn_s_wait_tensorcnt(1);
        else             __builtin_amdgcn_s_wait_tensorcnt(0);

        const __bf16* bp = &ldsA[wv][buf][0];
#pragma unroll
        for (int t = 0; t < 4; ++t) {
            const int row = t * 16 + (lane & 15);
            const __bf16* rp = bp + (size_t)row * 32 + half * 8;
            v8bf lo = *(const v8bf*)(rp);
            v8bf hi = *(const v8bf*)(rp + 16);
            v16bf af = __builtin_shufflevector(lo, hi, 0, 1, 2, 3, 4, 5, 6, 7,
                                                       8, 9, 10, 11, 12, 13, 14, 15);
            c[t][0] = WMMA_BF16(af, b0, c[t][0]);
            c[t][1] = WMMA_BF16(af, b1, c[t][1]);
        }
        buf ^= 1;
    }

#pragma unroll
    for (int t = 0; t < 4; ++t)
#pragma unroll
        for (int r = 0; r < 8; ++r) {
            int m = t * 16 + half * 8 + r;
            atomicAdd(&acc[(size_t)m * N1 + n],      c[t][0][r]);
            atomicAdd(&acc[(size_t)m * N1 + n + 16], c[t][1][r]);
        }
}

// ---------------------------------------------------------------------------
// Kernel 4: h1 = bf16(relu(acc + b1))     (64x512)
// ---------------------------------------------------------------------------
__global__ void k_bias_relu(const float* __restrict__ acc,
                            const float* __restrict__ b1,
                            __bf16* __restrict__ h1) {
    int i = blockIdx.x * blockDim.x + threadIdx.x;
    if (i < BATCH * N1)
        h1[i] = (__bf16)fmaxf(acc[i] + b1[i % N1], 0.0f);
}

// ---------------------------------------------------------------------------
// Kernel 5: GEMM2  h2 = bf16(relu(h1 @ W2 + b2))   (64x512)@(512x256)
// ---------------------------------------------------------------------------
__global__ void __launch_bounds__(32)
k_gemm2(const __bf16* __restrict__ h1, const float* __restrict__ W2,
        const float* __restrict__ b2, __bf16* __restrict__ h2) {
    const int lane  = threadIdx.x & 31;
    const int nbase = blockIdx.x * 16;

    v8f c[4];
#pragma unroll
    for (int t = 0; t < 4; ++t) c[t] = (v8f){0.f, 0.f, 0.f, 0.f, 0.f, 0.f, 0.f, 0.f};

    for (int k0 = 0; k0 < N1; k0 += 32) {
        v16bf bf = load_b_f32_as_bf16(W2, N2, k0, nbase, lane, N1);
#pragma unroll
        for (int t = 0; t < 4; ++t) {
            v16bf af = load_a_bf16(h1, N1, t * 16, k0, lane);
            c[t] = WMMA_BF16(af, bf, c[t]);
        }
    }

    const int half = lane >> 4;
    const int n    = nbase + (lane & 15);
    const float bias = b2[n];
#pragma unroll
    for (int t = 0; t < 4; ++t)
#pragma unroll
        for (int r = 0; r < 8; ++r) {
            int m = t * 16 + half * 8 + r;
            h2[(size_t)m * N2 + n] = (__bf16)fmaxf(c[t][r] + bias, 0.0f);
        }
}

// ---------------------------------------------------------------------------
// Kernel 6: GEMM3  logits = h2 @ Wo + bo  -> f32 (64x1008 padded)
// ---------------------------------------------------------------------------
__global__ void __launch_bounds__(32)
k_gemm3(const __bf16* __restrict__ h2, const float* __restrict__ Wo,
        const float* __restrict__ bo, float* __restrict__ logits) {
    const int lane  = threadIdx.x & 31;
    const int nbase = blockIdx.x * 16;

    v8f c[4];
#pragma unroll
    for (int t = 0; t < 4; ++t) c[t] = (v8f){0.f, 0.f, 0.f, 0.f, 0.f, 0.f, 0.f, 0.f};

    const int half = lane >> 4;
    int n  = nbase + (lane & 15);
    int nc = n < NOUT ? n : NOUT - 1;      // clamp reads in-bounds

    for (int k0 = 0; k0 < N2; k0 += 32) {
        v16bf bf;
#pragma unroll
        for (int e = 0; e < 16; ++e) {
            int k = k0 + half * 8 + (e & 7) + ((e >> 3) << 4);
            bf[e] = (__bf16)Wo[(size_t)k * NOUT + nc];
        }
#pragma unroll
        for (int t = 0; t < 4; ++t) {
            v16bf af = load_a_bf16(h2, N2, t * 16, k0, lane);
            c[t] = WMMA_BF16(af, bf, c[t]);
        }
    }

    const float bias = bo[nc];
#pragma unroll
    for (int t = 0; t < 4; ++t)
#pragma unroll
        for (int r = 0; r < 8; ++r) {
            int m = t * 16 + half * 8 + r;
            logits[(size_t)m * NOUTP + n] = c[t][r] + bias;
        }
}

// ---------------------------------------------------------------------------
// Kernel 7: row softmax over 1000 classes, LDS reductions. One block per row.
// ---------------------------------------------------------------------------
__global__ void __launch_bounds__(256)
k_softmax(const float* __restrict__ logits, float* __restrict__ out) {
    __shared__ float red[256];
    const int row = blockIdx.x;
    const int tid = threadIdx.x;
    const float* lr = logits + (size_t)row * NOUTP;

    float mx = -3.402823466e38f;
    for (int c = tid; c < NOUT; c += 256) mx = fmaxf(mx, lr[c]);
    red[tid] = mx; __syncthreads();
    for (int s = 128; s > 0; s >>= 1) {
        if (tid < s) red[tid] = fmaxf(red[tid], red[tid + s]);
        __syncthreads();
    }
    mx = red[0]; __syncthreads();

    float sum = 0.f;
    for (int c = tid; c < NOUT; c += 256) sum += __expf(lr[c] - mx);
    red[tid] = sum; __syncthreads();
    for (int s = 128; s > 0; s >>= 1) {
        if (tid < s) red[tid] += red[tid + s];
        __syncthreads();
    }
    const float inv = 1.0f / red[0];

    for (int c = tid; c < NOUT; c += 256)
        out[(size_t)row * NOUT + c] = __expf(lr[c] - mx) * inv;
}

// ---------------------------------------------------------------------------
// Host-side launcher (graph-capture safe: only kernel launches on `stream`)
// ---------------------------------------------------------------------------
extern "C" void kernel_launch(void* const* d_in, const int* in_sizes, int n_in,
                              void* d_out, int out_size, void* d_ws, size_t ws_size,
                              hipStream_t stream) {
    const float* x   = (const float*)d_in[0];
    const float* flt = (const float*)d_in[1];
    const float* W1  = (const float*)d_in[2];
    const float* b1  = (const float*)d_in[3];
    const float* W2  = (const float*)d_in[4];
    const float* b2  = (const float*)d_in[5];
    const float* Wo  = (const float*)d_in[6];
    const float* bo  = (const float*)d_in[7];
    float* out = (float*)d_out;

    // Workspace layout (all offsets 256B aligned)
    char* w = (char*)d_ws;
    __bf16* A      = (__bf16*)(w);                      // 64*98592*2 = 12,619,776
    float*  acc1   = (float*) (w + 12619776);           // 64*512*4   =    131,072
    __bf16* h1     = (__bf16*)(w + 12750848);           // 64*512*2   =     65,536
    __bf16* h2     = (__bf16*)(w + 12816384);           // 64*256*2   =     32,768
    float*  logits = (float*) (w + 12849152);           // 64*1008*4  =    258,048

    {
        int total = BATCH * KPAD;                        // 6,309,888
        k_conv_pool<<<dim3((total + 255) / 256), dim3(256), 0, stream>>>(x, flt, A);
    }
    k_zero<<<dim3((BATCH * N1 + 255) / 256), dim3(256), 0, stream>>>(acc1, BATCH * N1);
    k_gemm1<<<dim3(16, 16), dim3(256), 0, stream>>>(A, W1, acc1);
    k_bias_relu<<<dim3((BATCH * N1 + 255) / 256), dim3(256), 0, stream>>>(acc1, b1, h1);
    k_gemm2<<<dim3(N2 / 16), dim3(32), 0, stream>>>(h1, W2, b2, h2);
    k_gemm3<<<dim3(NOUTP / 16), dim3(32), 0, stream>>>(h2, Wo, bo, logits);
    k_softmax<<<dim3(BATCH), dim3(256), 0, stream>>>(logits, out);
}
